// AdaFace_85014582657458
// MI455X (gfx1250) — compile-verified, matrix operand order checked by reading, not run on hardware
//
#include <hip/hip_runtime.h>
#include <math.h>

typedef __attribute__((ext_vector_type(16))) _Float16 v16h;
typedef __attribute__((ext_vector_type(8)))  float    v8f;

#define N_ROWS   1024
#define DIM      512
#define NCLS     100000
#define CT       16            // classes per workgroup
#define LDSROW   (DIM + 8)     // padded LDS row stride (halves)
#define LDS_LO   (CT * LDSROW) // whi -> wlo offset (halves)
#define ALO_H    524288        // ahi -> alo offset in halves (1 MB)
#define ROWT_H   (16 * DIM)    // one row-tile stride in halves (16 KB)
#define MARGIN   0.4f
#define HSLOPE   0.333f
#define SCALE    64.0f
#define BN_MEAN  20.0f
#define BN_VARE  10000.001f    // BN_VAR + BN_EPS
#define PI_F     3.14159265358979f

union Frag { v16h h; uint4 u[2]; };

__device__ __forceinline__ float dot4(float4 a, float4 b) {
    return a.x * b.x + a.y * b.y + a.z * b.z + a.w * b.w;
}

// ---------------------------------------------------------------------------
// Kernel A: per-row norm, margin scaler g, normalized embs split into f16
// hi/lo parts (split-f16 GEMM trick). One wave (32 lanes) per row.
// Layout in d_ws: hi plane at abase, lo plane at abase + ALO_H halves.
// ---------------------------------------------------------------------------
__global__ __launch_bounds__(256) void prep_kernel(
    const float* __restrict__ embs, _Float16* __restrict__ abase,
    float* __restrict__ g_arr, float* __restrict__ rnorm_arr,
    float* __restrict__ rowsum) {
    const int wave = threadIdx.x >> 5;
    const int lane = threadIdx.x & 31;
    const int row  = blockIdx.x * 8 + wave;

    const float* e = embs + (size_t)row * DIM + lane * 16;
    float4 v[4];
    float ss = 0.0f;
#pragma unroll
    for (int q = 0; q < 4; ++q) {
        v[q] = ((const float4*)e)[q];
        ss += dot4(v[q], v[q]);
    }
#pragma unroll
    for (int m = 1; m < 32; m <<= 1) ss += __shfl_xor(ss, m, 32);

    const float norm = sqrtf(ss);
    const float rn   = 1.0f / norm;
    if (lane == 0) {
        float g = (norm - BN_MEAN) * (HSLOPE / sqrtf(BN_VARE));
        g = fminf(fmaxf(g, -1.0f), 1.0f);
        g_arr[row]     = g;
        rnorm_arr[row] = rn;
        rowsum[row]    = 0.0f;   // init softmax accumulator
    }

    Frag fh, fl;
#pragma unroll
    for (int q = 0; q < 4; ++q) {
        float vals[4] = {v[q].x, v[q].y, v[q].z, v[q].w};
#pragma unroll
        for (int u = 0; u < 4; ++u) {
            float x = vals[u] * rn;
            _Float16 h = (_Float16)x;
            fh.h[q * 4 + u] = h;
            fl.h[q * 4 + u] = (_Float16)(x - (float)h);
        }
    }
    _Float16* dst = abase + (size_t)row * DIM + lane * 16;
    ((uint4*)dst)[0]           = fh.u[0];
    ((uint4*)dst)[1]           = fh.u[1];
    ((uint4*)(dst + ALO_H))[0] = fl.u[0];
    ((uint4*)(dst + ALO_H))[1] = fl.u[1];
}

// ---------------------------------------------------------------------------
// Kernel B: main GEMM + fused shifted-softmax accumulation.
// One workgroup = CT(=16) classes x all 1024 rows. 8 waves; each wave owns 8
// row tiles, processed as 2 groups of 4 with 4 live accumulators so each
// B fragment (LDS) is reused across 4 row tiles. All A/B addresses use one
// base pointer + compile-time instruction offsets.
// Split-f16: logits = Ahi*Bhi + Ahi*Blo + Alo*Bhi (f32 accumulate).
// ---------------------------------------------------------------------------
__global__ __launch_bounds__(256) void gemm_kernel(
    const float* __restrict__ weight, const _Float16* __restrict__ abase,
    float* __restrict__ rowsum) {
    // [0] = hi plane, [1] = lo plane (contiguous -> constant DS offsets)
    __shared__ alignas(16) _Float16 wtile[2][CT][LDSROW];

    const int tid  = threadIdx.x;
    const int wave = tid >> 5;
    const int lane = tid & 31;
    const int cls_base = blockIdx.x * CT;

    // --- cooperative load + normalize + hi/lo split of the weight tile ---
    {
        const int c = tid >> 4;   // class within tile (0..15)
        const int j = tid & 15;   // 32-float chunk within the 512-dim row
        const float* wp = weight + (size_t)(cls_base + c) * DIM + j * 32;
        float4 r[8];
        float ss = 0.0f;
#pragma unroll
        for (int q = 0; q < 8; ++q) {
            r[q] = ((const float4*)wp)[q];
            ss += dot4(r[q], r[q]);
        }
        // 16 threads of one class live in the same wave half -> shfl reduce
#pragma unroll
        for (int m = 1; m < 16; m <<= 1) ss += __shfl_xor(ss, m, 32);
        const float rn = 1.0f / sqrtf(ss);
#pragma unroll
        for (int q = 0; q < 8; ++q) {
            float vals[4] = {r[q].x, r[q].y, r[q].z, r[q].w};
#pragma unroll
            for (int u = 0; u < 4; ++u) {
                float x = vals[u] * rn;
                _Float16 h = (_Float16)x;
                wtile[0][c][j * 32 + q * 4 + u] = h;
                wtile[1][c][j * 32 + q * 4 + u] = (_Float16)(x - (float)h);
            }
        }
    }
    __syncthreads();

    const int mcol = lane & 15;                 // output column / LDS class row
    const int koff = (lane < 16) ? 0 : 8;       // ISA f16 A/B fragment K split

    for (int grp = 0; grp < 2; ++grp) {
        const int rt0 = wave * 8 + grp * 4;     // first of 4 row tiles
        const _Float16* p = abase + (size_t)(rt0 * 16 + mcol) * DIM + koff;
        const _Float16* q = &wtile[0][mcol][koff];

        v8f acc[4] = {{}, {}, {}, {}};
#pragma unroll 2
        for (int kc = 0; kc < DIM; kc += 32) {
            Frag fbh, fbl, fah[4], fal[4];
            fbh.u[0] = *(const uint4*)(q);
            fbh.u[1] = *(const uint4*)(q + 16);
            fbl.u[0] = *(const uint4*)(q + LDS_LO);
            fbl.u[1] = *(const uint4*)(q + LDS_LO + 16);
#pragma unroll
            for (int r = 0; r < 4; ++r) {
                fah[r].u[0] = *(const uint4*)(p + r * ROWT_H);
                fah[r].u[1] = *(const uint4*)(p + r * ROWT_H + 16);
                fal[r].u[0] = *(const uint4*)(p + ALO_H + r * ROWT_H);
                fal[r].u[1] = *(const uint4*)(p + ALO_H + r * ROWT_H + 16);
            }
#pragma unroll
            for (int r = 0; r < 4; ++r) {
                acc[r] = __builtin_amdgcn_wmma_f32_16x16x32_f16(
                    false, fah[r].h, false, fbh.h, (short)0, acc[r], false, false);
                acc[r] = __builtin_amdgcn_wmma_f32_16x16x32_f16(
                    false, fah[r].h, false, fbl.h, (short)0, acc[r], false, false);
                acc[r] = __builtin_amdgcn_wmma_f32_16x16x32_f16(
                    false, fal[r].h, false, fbh.h, (short)0, acc[r], false, false);
            }
            p += 32;
            q += 32;
        }

        // clip -> scale -> exp with fixed shift of SCALE (logits < SCALE), then
        // per-row sums over the 16 columns (lanes of each half-wave) + atomic.
#pragma unroll
        for (int r = 0; r < 4; ++r) {
            float s[8];
#pragma unroll
            for (int v = 0; v < 8; ++v) {
                float l = fminf(fmaxf(acc[r][v], -0.999f), 0.999f) * SCALE;
                s[v] = __expf(l - SCALE);
            }
#pragma unroll
            for (int m = 1; m < 16; m <<= 1) {
#pragma unroll
                for (int v = 0; v < 8; ++v) s[v] += __shfl_xor(s[v], m, 32);
            }
            if (mcol == 0) {
                const int rbase = (rt0 + r) * 16 + ((lane >= 16) ? 8 : 0);
#pragma unroll
                for (int v = 0; v < 8; ++v) atomicAdd(&rowsum[rbase + v], s[v]);
            }
        }
    }
}

// ---------------------------------------------------------------------------
// Kernel C: exact fp32 label logit + AdaFace margins. One wave per row.
// ---------------------------------------------------------------------------
__global__ __launch_bounds__(256) void label_kernel(
    const float* __restrict__ embs, const float* __restrict__ weight,
    const int* __restrict__ labels, const float* __restrict__ g_arr,
    const float* __restrict__ rnorm_arr,
    float* __restrict__ lmod_arr, float* __restrict__ delta_arr) {
    const int wave = threadIdx.x >> 5;
    const int lane = threadIdx.x & 31;
    const int row  = blockIdx.x * 8 + wave;
    const int lab  = labels[row];

    const float* e = embs   + (size_t)row * DIM + lane * 16;
    const float* w = weight + (size_t)lab * DIM + lane * 16;
    float d = 0.0f, ws = 0.0f;
#pragma unroll
    for (int q = 0; q < 4; ++q) {
        float4 ev = ((const float4*)e)[q];
        float4 wv = ((const float4*)w)[q];
        d  += dot4(ev, wv);
        ws += dot4(wv, wv);
    }
#pragma unroll
    for (int m = 1; m < 32; m <<= 1) {
        d  += __shfl_xor(d, m, 32);
        ws += __shfl_xor(ws, m, 32);
    }
    if (lane == 0) {
        const float cosv = d * rnorm_arr[row] * (1.0f / sqrtf(ws));
        const float lc   = fminf(fmaxf(cosv, -0.999f), 0.999f);
        const float g    = g_arr[row];
        float th = acosf(lc) - MARGIN * g;          // angular margin
        th = fminf(fmaxf(th, 0.0f), PI_F);
        const float lm = (cosf(th) - (1.0f + MARGIN) * g) * SCALE; // + additive
        const float lo = lc * SCALE;
        lmod_arr[row]  = lm;
        // replace original label term in the exp-sum with the margined one
        delta_arr[row] = __expf(lm - SCALE) - __expf(lo - SCALE);
    }
}

// ---------------------------------------------------------------------------
// Kernel D: loss_n = log(sum_n + delta_n) + SCALE - lmod_n ; mean over rows.
// ---------------------------------------------------------------------------
__global__ __launch_bounds__(1024) void reduce_kernel(
    const float* __restrict__ rowsum, const float* __restrict__ delta,
    const float* __restrict__ lmod, float* __restrict__ out) {
    __shared__ float sbuf[32];
    const int t = threadIdx.x;
    float val = logf(rowsum[t] + delta[t]) + SCALE - lmod[t];
#pragma unroll
    for (int m = 1; m < 32; m <<= 1) val += __shfl_xor(val, m, 32);
    if ((t & 31) == 0) sbuf[t >> 5] = val;
    __syncthreads();
    if (t < 32) {
        float v = sbuf[t];
#pragma unroll
        for (int m = 1; m < 32; m <<= 1) v += __shfl_xor(v, m, 32);
        if (t == 0) out[0] = v * (1.0f / (float)N_ROWS);
    }
}

extern "C" void kernel_launch(void* const* d_in, const int* in_sizes, int n_in,
                              void* d_out, int out_size, void* d_ws, size_t ws_size,
                              hipStream_t stream) {
    (void)in_sizes; (void)n_in; (void)out_size; (void)ws_size;
    const float* embs   = (const float*)d_in[0];
    const float* weight = (const float*)d_in[1];
    const int*   labels = (const int*)d_in[2];
    float* out = (float*)d_out;

    char* ws = (char*)d_ws;
    _Float16* abase  = (_Float16*)(ws);                  // hi: 1 MB, lo: +1 MB
    float*    rowsum = (float*)(ws + (size_t)2097152);   // 4 KB
    float*    g_arr  = rowsum + N_ROWS;
    float*    rnorm  = g_arr  + N_ROWS;
    float*    lmod   = rnorm  + N_ROWS;
    float*    delta  = lmod   + N_ROWS;

    prep_kernel  <<<N_ROWS / 8, 256, 0, stream>>>(embs, abase, g_arr, rnorm, rowsum);
    gemm_kernel  <<<NCLS / CT,  256, 0, stream>>>(weight, abase, rowsum);
    label_kernel <<<N_ROWS / 8, 256, 0, stream>>>(embs, weight, labels, g_arr, rnorm,
                                                  lmod, delta);
    reduce_kernel<<<1, 1024, 0, stream>>>(rowsum, delta, lmod, out);
}
